// LatentDenoiser_75222057222561
// MI455X (gfx1250) — matrix operand order repaired and checked by reading, not compiled, not run on hardware
//
#include <hip/hip_runtime.h>
#include <hip/hip_bf16.h>
#include <math.h>

// ---------------------------------------------------------------------------
// LatentDenoiser fused inference for MI455X (gfx1250, wave32, WMMA).
//
// Compute-bound (0.22 TFLOP vs ~10MB HBM traffic) -> WMMA f16 GEMMs with f32
// accumulation. Each wave32 owns 16 batch rows and executes the full network:
// activations held as 16x16 WMMA C-fragments in registers, LayerNorm stats via
// shfl_xor butterflies, C->A layout transposes through a wave-private LDS tile
// (no barriers needed: per-wave LDS ops are in-order via DScnt).
//
// Input flattening assumption (setup_inputs() insertion order, recursive):
//  [0]z [1]t [2]y [3]t1W [4]t1b [5]t2W [6]t2b [7]class_emb [8]in_W [9]in_b
//  [10]out_norm_g [11]out_norm_b [12]out_W [13]out_b
//  [14+6j+{0..5}] = block j {norm_g, norm_b, proj_W, proj_b, fc_W, fc_b}
// Workspace assumption: ws_size >= ~3.5 MB (f16 weights + packed f32 biases).
// ---------------------------------------------------------------------------

typedef __attribute__((ext_vector_type(16))) _Float16 v16h;
typedef __attribute__((ext_vector_type(8)))  _Float16 v8h;
typedef __attribute__((ext_vector_type(8)))  float    v8f;

#define B_TOTAL 65536
#define HID 256

// f16 weight layout inside d_ws (offsets in halves)
#define OFF_T1W   0u          // 256 x 128
#define OFF_T2W   32768u      // 256 x 256
#define OFF_INW   98304u      // 256 x 32 (K padded 16->32 with zeros)
#define OFF_OUTW  106496u     // 16  x 256
#define OFF_BLK   110592u     // 8 x (proj 512x256, fc 256x256)
#define BLK_W_STRIDE 196608u
#define F16_TOTAL_HALVES 1683456u
#define F32_BIAS_OFF_FLOATS (F16_TOTAL_HALVES / 2u)   // byte offset /4

// ---------------------------------------------------------------------------
// prep kernels
// ---------------------------------------------------------------------------
__global__ void k_cast_f16(const float* __restrict__ src,
                           _Float16* __restrict__ dst, int n) {
  int i = blockIdx.x * 256 + threadIdx.x;
  if (i < n) dst[i] = (_Float16)src[i];
}

__global__ void k_pad_inw(const float* __restrict__ src,   // 256x16
                          _Float16* __restrict__ dst) {    // 256x32
  int i = blockIdx.x * 256 + threadIdx.x;
  if (i < 256 * 32) {
    int r = i >> 5, c = i & 31;
    dst[i] = (c < 16) ? (_Float16)src[r * 16 + c] : (_Float16)0.0f;
  }
}

__global__ void k_pack_bias(const float* __restrict__ ng, const float* __restrict__ nb,
                            const float* __restrict__ pb, const float* __restrict__ fb,
                            float* __restrict__ dst) {     // 1280 floats
  int i = blockIdx.x * 256 + threadIdx.x;
  if (i < 256)        dst[i] = ng[i];
  else if (i < 512)   dst[i] = nb[i - 256];
  else if (i < 1024)  dst[i] = pb[i - 512];
  else if (i < 1280)  dst[i] = fb[i - 1024];
}

// ---------------------------------------------------------------------------
// fragment helpers (wave32 WMMA 16x16x32 f16)
// ---------------------------------------------------------------------------
__device__ __forceinline__ v8f wmma32(v16h a, v16h b, v8f c) {
  // (neg_a, A, neg_b, B, c_mod, C, reuse_a, reuse_b)
  return __builtin_amdgcn_wmma_f32_16x16x32_f16(false, a, false, b,
                                                (short)0, c, false, false);
}

// B fragment: W is row-major (N x Kdim) f16. lane: n_local = lane&15, g = lane>>4.
// B[k][n] with k = c*32 + (e&7) + 8g + 16*(e>>3) (matches A layout).
__device__ __forceinline__ v16h ldB(const _Float16* __restrict__ W, int Kdim,
                                    int nrow, int c, int g) {
  const _Float16* p = W + (size_t)nrow * Kdim + c * 32 + g * 8;
  v8h lo = *(const v8h*)(p);
  v8h hi = *(const v8h*)(p + 16);
  v16h r;
#pragma unroll
  for (int i = 0; i < 8; i++) { r[i] = lo[i]; r[i + 8] = hi[i]; }
  return r;
}

// A fragment from row-major 16x256 f16 LDS tile. lane: m = lane&15, g = lane>>4.
// element e: k = c*32 + (e&7) + 8g + 16*(e>>3)  (ISA 16-bit A layout)
__device__ __forceinline__ v16h ldA(const _Float16* S, int c, int m, int g) {
  const _Float16* p = S + m * 256 + c * 32 + g * 8;
  v8h lo = *(const v8h*)(p);
  v8h hi = *(const v8h*)(p + 16);
  v16h r;
#pragma unroll
  for (int i = 0; i < 8; i++) { r[i] = lo[i]; r[i + 8] = hi[i]; }
  return r;
}

// SiLU with hardware rcp: x * v_rcp_f32(1 + exp(-x)); ~1 ulp rcp is fine here.
__device__ __forceinline__ float silu_f(float x) {
  return x * __builtin_amdgcn_rcpf(1.0f + __expf(-x));
}

struct DP {
  const float* z; const int* t; const int* y;
  const float* t1b; const float* t2b; const float* cemb;
  const float* inb;
  const float* ong; const float* onb; const float* ob;
  const _Float16* w16; const float* bias32;
  float* out;
};

// ---------------------------------------------------------------------------
// main fused kernel: 4 waves per block, 16 rows per wave
// ---------------------------------------------------------------------------
__global__ __launch_bounds__(128) void k_denoise(DP p) {
  __shared__ _Float16 smem[4 * 16 * 256];   // 32 KB: one 16x256 tile per wave
  const int lane = threadIdx.x & 31;
  const int wave = threadIdx.x >> 5;
  const int nl = lane & 15;     // C-layout: column; A-layout: row
  const int g  = lane >> 4;     // lane half
  const int row0 = (blockIdx.x * 4 + wave) * 16;
  _Float16* stage = smem + wave * (16 * 256);

  // ---------------- timestep embedding -> A fragments (16 x 128) ----------
  const float tf = (float)p.t[row0 + nl];
  v16h peA[4];
#pragma unroll
  for (int c = 0; c < 4; c++) {
#pragma unroll
    for (int e = 0; e < 16; e++) {
      int k = c * 32 + (e & 7) + 8 * g + 16 * (e >> 3);
      int j = (k < 64) ? k : k - 64;
      // freq = 10000^(-j/64) = 2^(-log2(1e4)*j/64); multiplier folds at
      // compile time (j literal), v_exp_f32 is natively exp2.
      float fr = exp2f(-13.28771238f * (float)j * (1.0f / 64.0f));
      float ang = tf * fr;
      peA[c][e] = (_Float16)((k < 64) ? __sinf(ang) : __cosf(ang));
    }
  }

  // ---------------- e1 = silu(pe @ t1W^T + t1b) -> stage ------------------
#pragma unroll
  for (int nt = 0; nt < 16; ++nt) {
    v8f acc = {};
#pragma unroll
    for (int c = 0; c < 4; c++)
      acc = wmma32(peA[c], ldB(p.w16 + OFF_T1W, 128, nt * 16 + nl, c, g), acc);
    int n = nt * 16 + nl;
    float bias = p.t1b[n];
#pragma unroll
    for (int r = 0; r < 8; r++)
      stage[(r + 8 * g) * 256 + n] = (_Float16)silu_f(acc[r] + bias);
  }

  // ---------------- e = silu(e1 @ t2W^T + t2b) + class_emb[y] -------------
  v16h eA[8];
#pragma unroll
  for (int c = 0; c < 8; c++) eA[c] = ldA(stage, c, nl, g);
  int ym[8];
#pragma unroll
  for (int r = 0; r < 8; r++) ym[r] = p.y[row0 + r + 8 * g];
#pragma unroll
  for (int nt = 0; nt < 16; ++nt) {
    v8f acc = {};
#pragma unroll
    for (int c = 0; c < 8; c++)
      acc = wmma32(eA[c], ldB(p.w16 + OFF_T2W, 256, nt * 16 + nl, c, g), acc);
    int n = nt * 16 + nl;
    float bias = p.t2b[n];
#pragma unroll
    for (int r = 0; r < 8; r++) {
      float v = silu_f(acc[r] + bias) + p.cemb[ym[r] * 256 + n];
      stage[(r + 8 * g) * 256 + n] = (_Float16)v;
    }
  }
#pragma unroll
  for (int c = 0; c < 8; c++) eA[c] = ldA(stage, c, nl, g);  // persistent e

  // ---------------- h = silu(z @ inW^T + inb) -----------------------------
  v16h zA;
#pragma unroll
  for (int e = 0; e < 16; e++) {
    int k = (e & 7) + 8 * g + 16 * (e >> 3);
    zA[e] = (k < 16) ? (_Float16)p.z[(size_t)(row0 + nl) * 16 + k]
                     : (_Float16)0.0f;
  }
  v8f H[16];
#pragma unroll
  for (int nt = 0; nt < 16; ++nt) {
    v8f acc = {};
    acc = wmma32(zA, ldB(p.w16 + OFF_INW, 32, nt * 16 + nl, 0, g), acc);
    int n = nt * 16 + nl;
    float bias = p.inb[n];
#pragma unroll
    for (int r = 0; r < 8; r++) H[nt][r] = silu_f(acc[r] + bias);
  }

  // ---------------- 4 residual groups x 2 FiLM blocks ---------------------
  v8f R[16];
#pragma unroll 1
  for (int grp = 0; grp < 4; ++grp) {
#pragma unroll
    for (int nt = 0; nt < 16; ++nt) R[nt] = H[nt];
#pragma unroll 1
    for (int bi = 0; bi < 2; ++bi) {
      const int blk = grp * 2 + bi;
      const _Float16* projW = p.w16 + OFF_BLK + (size_t)blk * BLK_W_STRIDE;
      const _Float16* fcW   = projW + 512 * 256;
      const float* bb = p.bias32 + blk * 1280;
      const float *ng = bb, *nb = bb + 256, *pb = bb + 512, *fb = bb + 1024;

      // layernorm stats (rows r+8g of this tile)
      float sum[8], ssq[8];
#pragma unroll
      for (int r = 0; r < 8; r++) { sum[r] = 0.f; ssq[r] = 0.f; }
#pragma unroll
      for (int nt = 0; nt < 16; ++nt)
#pragma unroll
        for (int r = 0; r < 8; r++) {
          float x = H[nt][r]; sum[r] += x; ssq[r] += x * x;
        }
#pragma unroll
      for (int m = 1; m < 16; m <<= 1)
#pragma unroll
        for (int r = 0; r < 8; r++) {
          sum[r] += __shfl_xor(sum[r], m, 32);
          ssq[r] += __shfl_xor(ssq[r], m, 32);
        }
      float mean[8], rsig[8];
#pragma unroll
      for (int r = 0; r < 8; r++) {
        mean[r] = sum[r] * (1.0f / 256.0f);
        float var = ssq[r] * (1.0f / 256.0f) - mean[r] * mean[r];
        rsig[r] = __builtin_amdgcn_rsqf(var + 1e-5f);
      }

      // gb = e @ projW^T; hmid = silu((1+gam)*LN(h) + bet) -> stage (f16)
#pragma unroll
      for (int nt = 0; nt < 16; ++nt) {
        v8f ga = {}, be = {};
#pragma unroll
        for (int c = 0; c < 8; c++) {
          ga = wmma32(eA[c], ldB(projW, 256, nt * 16 + nl, c, g), ga);
          be = wmma32(eA[c], ldB(projW, 256, 256 + nt * 16 + nl, c, g), be);
        }
        int n = nt * 16 + nl;
        float gb0 = pb[n], bb0 = pb[256 + n];
        float gn = ng[n], bnv = nb[n];
#pragma unroll
        for (int r = 0; r < 8; r++) {
          float xn = (H[nt][r] - mean[r]) * rsig[r] * gn + bnv;
          float hm = silu_f((1.0f + ga[r] + gb0) * xn + (be[r] + bb0));
          stage[(r + 8 * g) * 256 + n] = (_Float16)hm;
        }
      }

      // h = hmid @ fcW^T + fc_b   (per-wave LDS: in-order, no barrier)
      v16h hA[8];
#pragma unroll
      for (int c = 0; c < 8; c++) hA[c] = ldA(stage, c, nl, g);
#pragma unroll
      for (int nt = 0; nt < 16; ++nt) {
        v8f acc = {};
#pragma unroll
        for (int c = 0; c < 8; c++)
          acc = wmma32(hA[c], ldB(fcW, 256, nt * 16 + nl, c, g), acc);
        float bias = fb[nt * 16 + nl];
#pragma unroll
        for (int r = 0; r < 8; r++) H[nt][r] = acc[r] + bias;
      }
    }
#pragma unroll
    for (int nt = 0; nt < 16; ++nt)
#pragma unroll
      for (int r = 0; r < 8; r++) H[nt][r] += R[nt][r];
  }

  // ---------------- out = silu(LN(h)) @ outW^T + out_b --------------------
  {
    float sum[8], ssq[8];
#pragma unroll
    for (int r = 0; r < 8; r++) { sum[r] = 0.f; ssq[r] = 0.f; }
#pragma unroll
    for (int nt = 0; nt < 16; ++nt)
#pragma unroll
      for (int r = 0; r < 8; r++) {
        float x = H[nt][r]; sum[r] += x; ssq[r] += x * x;
      }
#pragma unroll
    for (int m = 1; m < 16; m <<= 1)
#pragma unroll
      for (int r = 0; r < 8; r++) {
        sum[r] += __shfl_xor(sum[r], m, 32);
        ssq[r] += __shfl_xor(ssq[r], m, 32);
      }
    float mean[8], rsig[8];
#pragma unroll
    for (int r = 0; r < 8; r++) {
      mean[r] = sum[r] * (1.0f / 256.0f);
      float var = ssq[r] * (1.0f / 256.0f) - mean[r] * mean[r];
      rsig[r] = __builtin_amdgcn_rsqf(var + 1e-5f);
    }
#pragma unroll
    for (int nt = 0; nt < 16; ++nt) {
      int n = nt * 16 + nl;
      float gn = p.ong[n], bnv = p.onb[n];
#pragma unroll
      for (int r = 0; r < 8; r++) {
        float xn = (H[nt][r] - mean[r]) * rsig[r] * gn + bnv;
        stage[(r + 8 * g) * 256 + n] = (_Float16)silu_f(xn);
      }
    }
    v16h oA[8];
#pragma unroll
    for (int c = 0; c < 8; c++) oA[c] = ldA(stage, c, nl, g);
    v8f acc = {};
#pragma unroll
    for (int c = 0; c < 8; c++)
      acc = wmma32(oA[c], ldB(p.w16 + OFF_OUTW, 256, nl, c, g), acc);
    float ob = p.ob[nl];
#pragma unroll
    for (int r = 0; r < 8; r++)
      p.out[(size_t)(row0 + r + 8 * g) * 16 + nl] = acc[r] + ob;
  }
}

// ---------------------------------------------------------------------------
extern "C" void kernel_launch(void* const* d_in, const int* in_sizes, int n_in,
                              void* d_out, int out_size, void* d_ws, size_t ws_size,
                              hipStream_t stream) {
  (void)in_sizes; (void)n_in; (void)out_size; (void)ws_size;

  _Float16* w16 = (_Float16*)d_ws;
  float* bias32 = (float*)d_ws + F32_BIAS_OFF_FLOATS;

  auto cast = [&](const void* src, size_t off, int n) {
    k_cast_f16<<<(n + 255) / 256, 256, 0, stream>>>((const float*)src, w16 + off, n);
  };

  // weight conversion (f32 -> f16) into workspace; deterministic each call
  cast(d_in[3],  OFF_T1W,  256 * 128);
  cast(d_in[5],  OFF_T2W,  256 * 256);
  cast(d_in[12], OFF_OUTW, 16 * 256);
  k_pad_inw<<<32, 256, 0, stream>>>((const float*)d_in[8], w16 + OFF_INW);
  for (int j = 0; j < 8; j++) {
    int b = 14 + 6 * j;
    cast(d_in[b + 2], OFF_BLK + (size_t)j * BLK_W_STRIDE, 512 * 256);
    cast(d_in[b + 4], OFF_BLK + (size_t)j * BLK_W_STRIDE + 512 * 256, 256 * 256);
    k_pack_bias<<<5, 256, 0, stream>>>(
        (const float*)d_in[b + 0], (const float*)d_in[b + 1],
        (const float*)d_in[b + 3], (const float*)d_in[b + 5],
        bias32 + j * 1280);
  }

  DP p;
  p.z    = (const float*)d_in[0];
  p.t    = (const int*)d_in[1];
  p.y    = (const int*)d_in[2];
  p.t1b  = (const float*)d_in[4];
  p.t2b  = (const float*)d_in[6];
  p.cemb = (const float*)d_in[7];
  p.inb  = (const float*)d_in[9];
  p.ong  = (const float*)d_in[10];
  p.onb  = (const float*)d_in[11];
  p.ob   = (const float*)d_in[13];
  p.w16  = w16;
  p.bias32 = bias32;
  p.out  = (float*)d_out;

  // 65536 rows / (16 rows/wave * 4 waves/block) = 1024 blocks
  k_denoise<<<1024, 128, 0, stream>>>(p);
}
